// SimplePCNet_41386304864897
// MI455X (gfx1250) — matrix operand order repaired and checked by reading, not compiled, not run on hardware
//
#include <hip/hip_runtime.h>
#include <hip/hip_bf16.h>

#define NPTS 100000
#define SPAD 258
#define S3   (258*258*258)
#define CH   256
#define NK   27
#define MT   128     // rows per block tile
#define AST  264     // padded LDS row stride in f16 elems (528B -> conflict-free)
#define NTIL 8       // 8 x 16-wide cout tiles per wave = 128 couts

typedef _Float16 f16;
typedef _Float16 v8h  __attribute__((ext_vector_type(8)));
typedef _Float16 v16h __attribute__((ext_vector_type(16)));
typedef float    v8f  __attribute__((ext_vector_type(8)));
typedef int      v4i  __attribute__((ext_vector_type(4)));

#if __has_builtin(__builtin_amdgcn_global_load_async_to_lds_b128)
#define HAVE_ASYNC_LDS 1
typedef __attribute__((address_space(1))) v4i gv4i;   // global AS payload ptr
typedef __attribute__((address_space(3))) v4i sv4i;   // LDS AS payload ptr
#else
#define HAVE_ASYNC_LDS 0
#endif

__device__ __forceinline__ void async_fence_wg() {
#if HAVE_ASYNC_LDS
#if __has_builtin(__builtin_amdgcn_s_wait_asynccnt)
    __builtin_amdgcn_s_wait_asynccnt(0);
#else
    asm volatile("s_wait_asynccnt 0" ::: "memory");
#endif
#endif
    __syncthreads();
}

// ---------------- neighbor map via dense voxel hash grid ----------------
__global__ void k_grid_init(int* __restrict__ grid, int n) {
    int i = blockIdx.x * blockDim.x + threadIdx.x;
    if (i < n) grid[i] = -1;
}

__global__ void k_grid_scatter(const int* __restrict__ coords, int* __restrict__ grid) {
    int p = blockIdx.x * blockDim.x + threadIdx.x;
    if (p >= NPTS) return;
    int x = coords[p*3+0] + 1, y = coords[p*3+1] + 1, z = coords[p*3+2] + 1;
    grid[(x*SPAD + y)*SPAD + z] = p;
}

__global__ void k_neighbors(const int* __restrict__ coords, const int* __restrict__ grid,
                            int* __restrict__ idxw) {
    int t = blockIdx.x * blockDim.x + threadIdx.x;
    if (t >= NK * NPTS) return;
    int k = t / NPTS, p = t % NPTS;
    int dx = k/9 - 1, dy = (k/3)%3 - 1, dz = k%3 - 1;
    int x = coords[p*3+0] + 1 + dx;
    int y = coords[p*3+1] + 1 + dy;
    int z = coords[p*3+2] + 1 + dz;
    idxw[t] = grid[(x*SPAD + y)*SPAD + z];     // -1 == invalid
}

// ------------- weight convert + transpose: [27][ci][co] f32 -> [27][co][ci] f16 -------------
__global__ void k_wcvt(const float* __restrict__ W, f16* __restrict__ wt) {
    int t = blockIdx.x * blockDim.x + threadIdx.x;
    if (t >= NK * CH * CH) return;
    int k = t / (CH*CH), r = t % (CH*CH);
    int co = r / CH, ci = r % CH;
    wt[((size_t)k*CH + co)*CH + ci] = (f16)W[((size_t)k*CH + ci)*CH + co];
}

// ---------------- layer 1: Cin=2 -> 256, VALU ----------------
__global__ void k_layer1(const float* __restrict__ feats, const float* __restrict__ W1,
                         const int* __restrict__ idxw, f16* __restrict__ xout) {
    int p = blockIdx.x, c = threadIdx.x;
    float acc = 0.f;
    for (int k = 0; k < NK; ++k) {
        int j = idxw[k*NPTS + p];
        if (j >= 0) {
            float f0 = feats[j*2+0], f1 = feats[j*2+1];
            acc += f0 * W1[(k*2+0)*CH + c] + f1 * W1[(k*2+1)*CH + c];
        }
    }
    xout[(size_t)p*CH + c] = (f16)acc;
}

// ---- gather one 128x256 f16 A tile for offset k into an LDS buffer ----
__device__ __forceinline__ void stage_tile(const f16* __restrict__ xin,
                                           const int* __restrict__ idxw,
                                           f16* buf, int k, int rowBase,
                                           int npts, int tid) {
#pragma unroll
    for (int it = 0; it < 16; ++it) {
        int cid = it*256 + tid;             // 0..4095 16B chunks
        int r   = cid >> 5;                 // row 0..127
        int cc  = (cid & 31) * 8;           // f16 column 0..248
        int p   = rowBase + r;
        int j   = (p < npts) ? idxw[k*npts + p] : -1;
#if HAVE_ASYNC_LDS
        if (j >= 0) {
            __builtin_amdgcn_global_load_async_to_lds_b128(
                (gv4i*)(xin + (size_t)j*CH + cc),
                (sv4i*)(buf + r*AST + cc), 0, 0);
        } else {
            v8h z = {0,0,0,0,0,0,0,0};
            *(v8h*)(buf + r*AST + cc) = z;
        }
#else
        v8h val = {0,0,0,0,0,0,0,0};
        if (j >= 0) val = *(const v8h*)(xin + (size_t)j*CH + cc);
        *(v8h*)(buf + r*AST + cc) = val;
        if (k + 1 < NK && (cid & 3) == 0) {
            int jn = (p < npts) ? idxw[(k+1)*npts + p] : -1;
            if (jn >= 0) __builtin_prefetch(xin + (size_t)jn*CH + cc, 0, 1);
        }
#endif
    }
}

// ---------------- layers 2/3: 256 -> 256, WMMA f32_16x16x32_f16 ----------------
__global__ __launch_bounds__(256) void k_conv(const f16* __restrict__ xin,
                                              const f16* __restrict__ wt,
                                              const int* __restrict__ idxw,
                                              f16* __restrict__ xout, int npts) {
    extern __shared__ f16 As[];                 // 2 x (128 x 264) f16 = 135168 B
    const int tid = threadIdx.x;
    const int ln  = tid & 31;                   // wave32 lane
    const int w   = tid >> 5;                   // wave id (0..7)
    const int lo  = ln & 15;
    const int hi  = ln >> 4;
    const int rowBase  = blockIdx.x * MT;
    const int coutBase = blockIdx.y * 128;

    f16* buf0 = As;
    f16* buf1 = As + (size_t)MT * AST;

    v8f acc[NTIL];
#pragma unroll
    for (int i = 0; i < NTIL; ++i) acc[i] = (v8f){0.f,0.f,0.f,0.f,0.f,0.f,0.f,0.f};

    stage_tile(xin, idxw, buf0, 0, rowBase, npts, tid);
    async_fence_wg();

    for (int k = 0; k < NK; ++k) {
        f16* cur = (k & 1) ? buf1 : buf0;
        f16* nxt = (k & 1) ? buf0 : buf1;
        if (k + 1 < NK)                          // overlap next gather with compute
            stage_tile(xin, idxw, nxt, k + 1, rowBase, npts, tid);

        // per-lane bases; all inner offsets fold to instruction immediates
        const f16* arow = cur + (16*w + lo) * AST + hi * 16;
        const f16* wk   = wt + (size_t)k*CH*CH + (size_t)(coutBase + lo)*CH + hi*16;

#pragma unroll
        for (int ks = 0; ks < 8; ++ks) {
            union { v16h v; v8h h[2]; } af;      // A frag: row=lane%16, 16 contiguous K (same order as B)
            af.h[0] = *(const v8h*)(arow + ks*32);
            af.h[1] = *(const v8h*)(arow + ks*32 + 8);
#pragma unroll
            for (int nt = 0; nt < NTIL; ++nt) {  // B frag: col=lane%16, 16 contiguous K
                v16h bf = *(const v16h*)(wk + nt*(16*CH) + ks*32);
                acc[nt] = __builtin_amdgcn_wmma_f32_16x16x32_f16(
                    false, af.v, false, bf, (short)0, acc[nt], false, false);
            }
        }
        async_fence_wg();                        // next buffer ready, computes drained
    }

    // ---- store: C/D layout -> VGPR v holds M = v + 8*(lane/16), col = lane%16 ----
    f16* orow = xout + (size_t)(rowBase + 16*w + 8*hi) * CH + coutBase + lo;
#pragma unroll
    for (int nt = 0; nt < NTIL; ++nt) {
#pragma unroll
        for (int v = 0; v < 8; ++v) {
            int p = rowBase + 16*w + v + 8*hi;
            if (p < npts) orow[(size_t)v*CH + nt*16] = (f16)acc[nt][v];
        }
    }
}

// ---------------- layer 4: 256 -> 2, VALU dot + block reduce ----------------
__global__ void k_layer4(const f16* __restrict__ x, const float* __restrict__ W4,
                         const int* __restrict__ idxw, float* __restrict__ out) {
    __shared__ float s0[256], s1[256];
    int p = blockIdx.x, t = threadIdx.x;
    float a0 = 0.f, a1 = 0.f;
    for (int k = 0; k < NK; ++k) {
        int j = idxw[k*NPTS + p];
        if (j >= 0) {
            float xv = (float)x[(size_t)j*CH + t];
            a0 += xv * W4[(k*CH + t)*2 + 0];
            a1 += xv * W4[(k*CH + t)*2 + 1];
        }
    }
    s0[t] = a0; s1[t] = a1;
    __syncthreads();
    for (int s = 128; s > 0; s >>= 1) {
        if (t < s) { s0[t] += s0[t + s]; s1[t] += s1[t + s]; }
        __syncthreads();
    }
    if (t == 0) { out[p*2 + 0] = s0[0]; out[p*2 + 1] = s1[0]; }
}

// ---------------------------------------------------------------------------
extern "C" void kernel_launch(void* const* d_in, const int* in_sizes, int n_in,
                              void* d_out, int out_size, void* d_ws, size_t ws_size,
                              hipStream_t stream) {
    const int*   coords = (const int*)d_in[0];
    const float* feats  = (const float*)d_in[1];
    const float* W1     = (const float*)d_in[2];
    const float* W2     = (const float*)d_in[3];
    const float* W3     = (const float*)d_in[4];
    const float* W4     = (const float*)d_in[5];

    char* ws = (char*)d_ws;
    size_t off = 0;
    auto carve = [&](size_t bytes) -> char* {
        char* p = ws + off;
        off += (bytes + 255) & ~(size_t)255;
        return p;
    };
    int* grid = (int*)carve((size_t)S3 * sizeof(int));               // 68.7 MB
    int* idxw = (int*)carve((size_t)NK * NPTS * sizeof(int));        // 10.8 MB
    f16* x1   = (f16*)carve((size_t)NPTS * CH * sizeof(f16));        // 51.2 MB
    f16* x2   = (f16*)carve((size_t)NPTS * CH * sizeof(f16));        // 51.2 MB
    f16* wt2  = (f16*)carve((size_t)NK * CH * CH * sizeof(f16));     // 3.5 MB
    f16* wt3  = (f16*)carve((size_t)NK * CH * CH * sizeof(f16));     // 3.5 MB

    k_grid_init   <<<(S3 + 255)/256,        256, 0, stream>>>(grid, S3);
    k_grid_scatter<<<(NPTS + 255)/256,      256, 0, stream>>>(coords, grid);
    k_neighbors   <<<(NK*NPTS + 255)/256,   256, 0, stream>>>(coords, grid, idxw);
    k_wcvt        <<<(NK*CH*CH + 255)/256,  256, 0, stream>>>(W2, wt2);
    k_wcvt        <<<(NK*CH*CH + 255)/256,  256, 0, stream>>>(W3, wt3);

    k_layer1<<<NPTS, 256, 0, stream>>>(feats, W1, idxw, x1);

    dim3 cg((NPTS + MT - 1)/MT, 2);
    size_t ldsB = (size_t)2 * MT * AST * sizeof(f16);   // 135168 B dynamic LDS
    k_conv<<<cg, 256, ldsB, stream>>>(x1, wt2, idxw, x2, NPTS);
    k_conv<<<cg, 256, ldsB, stream>>>(x2, wt3, idxw, x1, NPTS);

    k_layer4<<<NPTS, 256, 0, stream>>>(x1, W4, idxw, (float*)d_out);
}
